// TernaryLinear_80994493268388
// MI455X (gfx1250) — matrix operand order repaired and checked by reading, not compiled, not run on hardware
//
#include <hip/hip_runtime.h>

typedef _Float16 f16;
typedef __attribute__((ext_vector_type(16))) _Float16 v16h;
typedef __attribute__((ext_vector_type(8)))  _Float16 v8h;
typedef __attribute__((ext_vector_type(4)))  _Float16 v4h;
typedef __attribute__((ext_vector_type(8)))  float    v8f;
typedef __attribute__((ext_vector_type(4)))  float    v4f;

#define M_DIM   8192      // B*S
#define K_DIM   4096      // IN
#define N_DIM   16384     // OUT

#define BM 128            // block tile M
#define BN 256            // block tile N
#define BK 32             // K per stage (= one WMMA K-step)
#define LDSH 40           // halves per LDS row (32 + 8 pad) = 80 B, 16B-aligned
#define A_OFF_H 0
#define B_OFF_H (BM * LDSH)                       // A tile: 128 rows
#define BUF_HALVES ((BM + BN) * LDSH)             // 15360 halves
#define BUF_BYTES  (BUF_HALVES * 2)               // 30720 B per buffer

// async copy of 16B per lane, global -> LDS; offset applies to BOTH addresses
#define ASYNC_CP16(lds, g, off)                                            \
  asm volatile("global_load_async_to_lds_b128 %0, %1, off offset:" #off    \
               :: "v"(lds), "v"(g) : "memory")

#define WAIT_ASYNC0() asm volatile("s_wait_asynccnt 0" ::: "memory")

// ---------------- phase 1a: x fp32 -> fp16 ----------------
__global__ __launch_bounds__(256) void cvt_x_kernel(const v4f* __restrict__ x,
                                                    v4h* __restrict__ xh, int n4) {
  int i = blockIdx.x * blockDim.x + threadIdx.x;
  if (i >= n4) return;
  v4f v = __builtin_nontemporal_load(&x[i]);
  v4h h = { (f16)v.x, (f16)v.y, (f16)v.z, (f16)v.w };
  xh[i] = h;
}

// ---------------- phase 1b: W fp32 -> ternary fp16 {-1,0,+1} ----------------
__device__ __forceinline__ f16 tq(float a) {
  return (f16)((a > 0.5f) ? 1.0f : ((a < -0.5f) ? -1.0f : 0.0f));
}

__global__ __launch_bounds__(256) void cvt_w_kernel(const v4f* __restrict__ w,
                                                    v4h* __restrict__ wh, int n4) {
  int i = blockIdx.x * blockDim.x + threadIdx.x;
  if (i >= n4) return;
  v4f v = __builtin_nontemporal_load(&w[i]);
  v4h h = { tq(v.x), tq(v.y), tq(v.z), tq(v.w) };
  wh[i] = h;
}

// ---------------- phase 2: fp16 WMMA GEMM, async-LDS double buffered ----------------
// C[M,N] = Xh[M,K] * Wh[N,K]^T * scale[N]
__global__ __launch_bounds__(256) void wmma_gemm_kernel(const f16* __restrict__ Xh,
                                                        const f16* __restrict__ Wh,
                                                        const float* __restrict__ scale,
                                                        float* __restrict__ Out) {
  __shared__ __attribute__((aligned(16))) f16 smem[2 * BUF_HALVES];   // 60 KB

  const int tid  = threadIdx.x;
  const int lane = tid & 31;
  const int wid  = tid >> 5;     // 0..7
  const int wrow = wid >> 2;     // 0..1 -> 64-row group
  const int wcol = wid & 3;      // 0..3 -> 64-col group

  // grid swizzle: stripes of 8 M-tiles; consecutive blocks share B panels
  const int NT = N_DIM / BN;     // 64
  const int GRP = 8;
  const int bid    = blockIdx.x;
  const int stripe = bid / (GRP * NT);
  const int within = bid % (GRP * NT);
  const int m0 = (stripe * GRP + (within % GRP)) * BM;
  const int n0 = (within / GRP) * BN;

  // global->LDS mapping: A: 2 thr/row, 16 halves each; B: 1 thr/row, 32 halves
  const int aRow = tid >> 1;
  const int aOff = (tid & 1) * 16;     // halves
  const int bRow = tid;

  const f16* xg = Xh + (size_t)(m0 + aRow) * K_DIM + aOff;
  const f16* wg = Wh + (size_t)(n0 + bRow) * K_DIM;

  const unsigned ldsBase = (unsigned)(uintptr_t)smem;  // low 32 bits = LDS byte addr
  const unsigned aSt = ldsBase + (A_OFF_H + aRow * LDSH + aOff) * 2;
  const unsigned bSt = ldsBase + (B_OFF_H + bRow * LDSH) * 2;

  // prologue: stage 0 -> buffer 0
  {
    unsigned long long ga = (unsigned long long)(uintptr_t)xg;
    unsigned long long gb = (unsigned long long)(uintptr_t)wg;
    ASYNC_CP16(aSt, ga, 0);  ASYNC_CP16(aSt, ga, 16);
    ASYNC_CP16(bSt, gb, 0);  ASYNC_CP16(bSt, gb, 16);
    ASYNC_CP16(bSt, gb, 32); ASYNC_CP16(bSt, gb, 48);
  }

  const int r  = lane & 15;   // A row / B column within 16
  const int hi = lane >> 4;   // lane-half selector

  v8f acc[4][4] = {};

  const int NSTAGE = K_DIM / BK;   // 128
  for (int t = 0; t < NSTAGE; ++t) {
    const int cur = t & 1;

    WAIT_ASYNC0();        // my stage-t loads landed
    __syncthreads();      // everyone's landed; everyone done reading buf(t+1)

    if (t + 1 < NSTAGE) { // prefetch stage t+1 into the other buffer
      const int nxt = cur ^ 1;
      unsigned long long ga = (unsigned long long)(uintptr_t)(xg + (size_t)(t + 1) * BK);
      unsigned long long gb = (unsigned long long)(uintptr_t)(wg + (size_t)(t + 1) * BK);
      unsigned aD = aSt + nxt * BUF_BYTES;
      unsigned bD = bSt + nxt * BUF_BYTES;
      ASYNC_CP16(aD, ga, 0);  ASYNC_CP16(aD, ga, 16);
      ASYNC_CP16(bD, gb, 0);  ASYNC_CP16(bD, gb, 16);
      ASYNC_CP16(bD, gb, 32); ASYNC_CP16(bD, gb, 48);
    }

    const f16* As = smem + cur * BUF_HALVES;
    const f16* Bs = smem + cur * BUF_HALVES + B_OFF_H;

    v16h a[4], b[4];
    // A fragment (16x32 f16): lane<16 -> row r, K {k0..k0+7, k0+16..k0+23}, k0=hi*8
#pragma unroll
    for (int i = 0; i < 4; ++i) {
      const f16* p = As + (wrow * 64 + i * 16 + r) * LDSH + hi * 8;
      v8h lo = *(const v8h*)p;
      v8h ho = *(const v8h*)(p + 16);
      a[i] = __builtin_shufflevector(lo, ho, 0, 1, 2, 3, 4, 5, 6, 7,
                                             8, 9, 10, 11, 12, 13, 14, 15);
    }
    // B fragment (32x16 f16): lane holds column r, K {k0..k0+7, k0+16..k0+23}, k0=hi*8
#pragma unroll
    for (int j = 0; j < 4; ++j) {
      const f16* p = Bs + (wcol * 64 + j * 16 + r) * LDSH + hi * 8;
      v8h lo = *(const v8h*)p;
      v8h ho = *(const v8h*)(p + 16);
      b[j] = __builtin_shufflevector(lo, ho, 0, 1, 2, 3, 4, 5, 6, 7,
                                             8, 9, 10, 11, 12, 13, 14, 15);
    }
#pragma unroll
    for (int i = 0; i < 4; ++i)
#pragma unroll
      for (int j = 0; j < 4; ++j)
        acc[i][j] = __builtin_amdgcn_wmma_f32_16x16x32_f16(
            false, a[i], false, b[j], (short)0, acc[i][j], false, false);
  }

  // ---- epilogue: per-channel scale, non-temporal fp32 stores (protect L2) ----
  // C/D layout: lane<16: N=lane, VGPR e -> M=e ; lane>=16: N=lane-16, VGPR e -> M=8+e
#pragma unroll
  for (int j = 0; j < 4; ++j) {
    const int n = n0 + wcol * 64 + j * 16 + r;
    const float s = scale[n];
#pragma unroll
    for (int i = 0; i < 4; ++i) {
      const int mb = m0 + wrow * 64 + i * 16 + hi * 8;
#pragma unroll
      for (int e = 0; e < 8; ++e)
        __builtin_nontemporal_store(acc[i][j][e] * s,
                                    &Out[(size_t)(mb + e) * N_DIM + n]);
    }
  }
}

extern "C" void kernel_launch(void* const* d_in, const int* in_sizes, int n_in,
                              void* d_out, int out_size, void* d_ws, size_t ws_size,
                              hipStream_t stream) {
  const float* x     = (const float*)d_in[0];   // [4,2048,4096] fp32
  const float* w     = (const float*)d_in[1];   // [16384,4096] fp32
  const float* scale = (const float*)d_in[2];   // [16384] fp32
  float* out = (float*)d_out;                   // [4,2048,16384] fp32

  // Workspace: xh (64 MiB) | wh (128 MiB) -> needs 192 MiB of d_ws
  f16* xh = (f16*)d_ws;
  f16* wh = (f16*)((char*)d_ws + (size_t)M_DIM * K_DIM * sizeof(f16));

  const int xn4 = M_DIM * K_DIM / 4;
  const int wn4 = N_DIM * K_DIM / 4;
  cvt_x_kernel<<<xn4 / 256, 256, 0, stream>>>((const v4f*)x, (v4h*)xh, xn4);
  cvt_w_kernel<<<wn4 / 256, 256, 0, stream>>>((const v4f*)w, (v4h*)wh, wn4);

  const int nblocks = (M_DIM / BM) * (N_DIM / BN);   // 64 * 64 = 4096
  wmma_gemm_kernel<<<nblocks, 256, 0, stream>>>(xh, wh, scale, out);
}
